// SiameseModel_3298534884287
// MI455X (gfx1250) — matrix operand order repaired and checked by reading, not compiled, not run on hardware
//
#include <hip/hip_runtime.h>
#include <hip/hip_bf16.h>

// ---------------- problem constants ----------------
#define B_    512
#define T_    60
#define E_    300
#define H_    512
#define M_    256
#define FOURH 2048
#define EPAD  320                       // x-part K padded to multiple of 32
#define KTOT  832                       // EPAD + H
#define KT_   26                        // KTOT / 32  (K-tiles, even)
#define NTILES 128                      // FOURH / 16 (N-tiles)
#define FRAG_ELEMS 512                  // one 32x16 bf16 B-fragment
#define FRAG_TOTAL (KT_ * NTILES * FRAG_ELEMS)   // 1,703,936 bf16 elems
#define ASTRIDE 848                     // 832 + 16 pad (keeps 16B align, spreads banks)
#define COMB  2053                      // 4*(H+1)+1

typedef __attribute__((ext_vector_type(16))) __bf16 v16bf;
typedef __attribute__((ext_vector_type(8)))  __bf16 bf16x8;
typedef __attribute__((ext_vector_type(8)))  float  v8f;

union V16 { v16bf v; bf16x8 h[2]; };

__device__ __forceinline__ float sigm(float x) { return 1.0f / (1.0f + __expf(-x)); }

// ---------------------------------------------------------------------------
// Kernel 0: pack lstm_kernel [812, 2048] f32 -> bf16 B-fragments in WMMA lane
// order. Fragment (kt, nt): lane l holds 16 bf16 contiguous at lane*16:
//   N = nt*16 + (l%16),  K = kt*32 + (l<16 ? 0 : 16) + j   (j = 0..15)
// K in [300,320) is zero padding; K >= 320 maps to kernel row 300 + (K-320).
// ---------------------------------------------------------------------------
__global__ void prep_wfrag(const float* __restrict__ kern, __bf16* __restrict__ frag) {
    int idx = blockIdx.x * blockDim.x + threadIdx.x;
    if (idx >= FRAG_TOTAL) return;
    int fragId = idx >> 9;          // /512
    int within = idx & 511;
    int lane   = within >> 4;
    int j      = within & 15;
    int kt = fragId >> 7;           // /128
    int nt = fragId & 127;
    int K = kt * 32 + ((lane & 16) ? 16 : 0) + j;
    int n = nt * 16 + (lane & 15);
    float v = 0.0f;
    if (K < E_)            v = kern[(size_t)K * FOURH + n];
    else if (K >= EPAD)    v = kern[(size_t)(K - EPAD + E_) * FOURH + n];
    frag[idx] = (__bf16)v;
}

// ---------------------------------------------------------------------------
// Kernel 1: persistent-per-16-rows LSTM. 64 blocks (2 seqs x 32 row chunks),
// 8 waves/block. Wave w owns hidden units [32w, 32w+32) in all 4 gates:
// 8 bf16 WMMA accumulators; c/h state lives in VGPRs across the time loop.
// B fragments are software-pipelined through two named register buffers so
// tile kt+1's global_load_b128s are in flight while tile kt's WMMAs execute.
// ---------------------------------------------------------------------------
__global__ __launch_bounds__(256) void lstm_kernel(
    const int* __restrict__ in1, const int* __restrict__ in2,
    const int* __restrict__ sl1, const int* __restrict__ sl2,
    const float* __restrict__ emb, const __bf16* __restrict__ frag,
    const float* __restrict__ bias, float* __restrict__ hbuf) {

    __shared__ __align__(16) __bf16 Atile[16 * ASTRIDE];

    const int seq   = blockIdx.x >> 5;
    const int chunk = blockIdx.x & 31;
    const int b0    = chunk * 16;
    const int* __restrict__ inp = seq ? in2 : in1;
    const int* __restrict__ sl  = seq ? sl2 : sl1;

    const int tid  = threadIdx.x;
    const int w    = tid >> 5;          // wave id 0..7
    const int lane = tid & 31;
    const int col  = lane & 15;         // N within tile
    const int hi   = lane >> 4;         // lane half

    // seqlens for the 8 M-rows this lane touches (row m = r + 8*hi)
    int slr[8];
    #pragma unroll
    for (int r = 0; r < 8; ++r) slr[r] = sl[b0 + r + 8 * hi];

    v8f cR[2], hR[2];
    #pragma unroll
    for (int s = 0; s < 2; ++s)
        #pragma unroll
        for (int r = 0; r < 8; ++r) { cR[s][r] = 0.0f; hR[s][r] = 0.0f; }

    // zero the h region of the A tile
    for (int idx = tid; idx < 16 * H_; idx += 256) {
        int row = idx >> 9; int u = idx & 511;
        Atile[row * ASTRIDE + EPAD + u] = (__bf16)0.0f;
    }

    // A-fragment base for this lane (row = col, K offset = hi*16)
    const __bf16* arow = &Atile[col * ASTRIDE + hi * 16];
    // B-fragment base for this wave/lane; tile (kt, g, s) is at
    // base + (kt*NTILES + g*32 + 2w + s) * FRAG_ELEMS
    const __bf16* bbase = frag + (size_t)(2 * w) * FRAG_ELEMS + lane * 16;

    for (int t = 0; t < T_; ++t) {
        // ---- fill x part of A tile (embedding gather, f32 -> bf16) ----
        for (int idx = tid; idx < 16 * EPAD; idx += 256) {
            int row = idx / EPAD; int e = idx - row * EPAD;
            int token = inp[(b0 + row) * T_ + t];
            float v = (e < E_) ? emb[(size_t)token * E_ + e] : 0.0f;
            Atile[row * ASTRIDE + e] = (__bf16)v;
        }
        __syncthreads();   // x + previous h visible to every wave

        // ---- GEMM: [16, 832] @ [832, 32 cols x 4 gates] for this wave ----
        v8f acc[4][2];
        #pragma unroll
        for (int g = 0; g < 4; ++g)
            #pragma unroll
            for (int s = 0; s < 2; ++s)
                #pragma unroll
                for (int r = 0; r < 8; ++r) acc[g][s][r] = 0.0f;

        V16 bA[4][2], bB[4][2];

        // prologue: load kt = 0 into buffer A
        #pragma unroll
        for (int g = 0; g < 4; ++g)
            #pragma unroll
            for (int s = 0; s < 2; ++s) {
                const __bf16* bp = bbase + (size_t)(g * 32 + s) * FRAG_ELEMS;
                bA[g][s].h[0] = *(const bf16x8*)bp;
                bA[g][s].h[1] = *(const bf16x8*)(bp + 8);
            }

        #pragma unroll 1
        for (int kt = 0; kt < KT_; kt += 2) {
            const __bf16* bk1 = bbase + (size_t)(kt + 1) * NTILES * FRAG_ELEMS;
            const __bf16* bk2 = bbase + (size_t)(kt + 2) * NTILES * FRAG_ELEMS;

            // ---- phase 0: preload kt+1 into B, compute kt from A ----
            V16 a0;
            a0.h[0] = *(const bf16x8*)(arow + kt * 32);
            a0.h[1] = *(const bf16x8*)(arow + kt * 32 + 8);
            __builtin_prefetch(bk2, 0, 3);   // warm near caches for kt+2
            #pragma unroll
            for (int g = 0; g < 4; ++g)
                #pragma unroll
                for (int s = 0; s < 2; ++s) {
                    const __bf16* bp = bk1 + (size_t)(g * 32 + s) * FRAG_ELEMS;
                    bB[g][s].h[0] = *(const bf16x8*)bp;
                    bB[g][s].h[1] = *(const bf16x8*)(bp + 8);
                }
            #pragma unroll
            for (int g = 0; g < 4; ++g)
                #pragma unroll
                for (int s = 0; s < 2; ++s)
                    acc[g][s] = __builtin_amdgcn_wmma_f32_16x16x32_bf16(
                        false, a0.v, false, bA[g][s].v, (short)0, acc[g][s],
                        false, false);

            // ---- phase 1: preload kt+2 into A, compute kt+1 from B ----
            V16 a1;
            a1.h[0] = *(const bf16x8*)(arow + (kt + 1) * 32);
            a1.h[1] = *(const bf16x8*)(arow + (kt + 1) * 32 + 8);
            if (kt + 2 < KT_) {
                #pragma unroll
                for (int g = 0; g < 4; ++g)
                    #pragma unroll
                    for (int s = 0; s < 2; ++s) {
                        const __bf16* bp = bk2 + (size_t)(g * 32 + s) * FRAG_ELEMS;
                        bA[g][s].h[0] = *(const bf16x8*)bp;
                        bA[g][s].h[1] = *(const bf16x8*)(bp + 8);
                    }
            }
            #pragma unroll
            for (int g = 0; g < 4; ++g)
                #pragma unroll
                for (int s = 0; s < 2; ++s)
                    acc[g][s] = __builtin_amdgcn_wmma_f32_16x16x32_bf16(
                        false, a1.v, false, bB[g][s].v, (short)0, acc[g][s],
                        false, false);
        }

        // ---- bias + gate math + seqlen masking (all in registers) ----
        #pragma unroll
        for (int s = 0; s < 2; ++s) {
            int u = 32 * w + 16 * s + col;
            float bi = bias[u];
            float bj = bias[512 + u];
            float bf_ = bias[1024 + u];
            float bo = bias[1536 + u];
            #pragma unroll
            for (int r = 0; r < 8; ++r) {
                float iv = acc[0][s][r] + bi;
                float jv = acc[1][s][r] + bj;
                float fv = acc[2][s][r] + bf_;
                float ov = acc[3][s][r] + bo;
                float nc = cR[s][r] * sigm(fv + 1.0f) + sigm(iv) * tanhf(jv);
                float nh = tanhf(nc) * sigm(ov);
                bool valid = t < slr[r];
                cR[s][r] = valid ? nc : cR[s][r];
                hR[s][r] = valid ? nh : hR[s][r];
            }
        }

        __syncthreads();   // everyone done reading A tile before h overwrite

        // ---- write new h slice into LDS A tile (bf16) ----
        #pragma unroll
        for (int s = 0; s < 2; ++s) {
            int u = 32 * w + 16 * s + col;
            #pragma unroll
            for (int r = 0; r < 8; ++r) {
                int m = r + 8 * hi;
                Atile[m * ASTRIDE + EPAD + u] = (__bf16)hR[s][r];
            }
        }
        // next iteration's x-fill touches a disjoint region; its barrier
        // publishes these h writes before the next GEMM.
    }

    // ---- final h (fp32) to workspace ----
    #pragma unroll
    for (int s = 0; s < 2; ++s) {
        int u = 32 * w + 16 * s + col;
        #pragma unroll
        for (int r = 0; r < 8; ++r) {
            int m = r + 8 * hi;
            hbuf[((size_t)seq * B_ + b0 + m) * H_ + u] = hR[s][r];
        }
    }
}

// ---------------------------------------------------------------------------
// Kernel 2: feature combine + MLP head (tiny: ~0.5 GFLOP). One block per row.
// ---------------------------------------------------------------------------
__global__ __launch_bounds__(256) void head_kernel(
    const float* __restrict__ hbuf, const int* __restrict__ sl1,
    const int* __restrict__ sl2, const float* __restrict__ W1,
    const float* __restrict__ b1, const float* __restrict__ W2,
    const float* __restrict__ b2, float* __restrict__ out) {

    __shared__ float comb[COMB];
    __shared__ float e1[M_];
    __shared__ float red[256];
    __shared__ float sdist;

    int b = blockIdx.x;
    int tid = threadIdx.x;
    if (tid == 0) sdist = 0.0f;
    __syncthreads();

    const float* h1 = hbuf + (size_t)b * H_;
    const float* h2 = hbuf + (size_t)(B_ + b) * H_;
    float l1 = (float)sl1[b] / (float)T_;
    float l2 = (float)sl2[b] / (float)T_;

    float part = 0.0f;
    for (int k = tid; k < 513; k += 256) {
        float a = (k < H_) ? h1[k] : l1;
        float c = (k < H_) ? h2[k] : l2;
        comb[k]        = a;        // h1
        comb[513 + k]  = c;        // h2
        float d = a - c;
        comb[1026 + k] = d;        // h_sub
        comb[1540 + k] = a * c;    // h_mul
        part += d * d;
    }
    atomicAdd(&sdist, part);
    __syncthreads();
    if (tid == 0) comb[1539] = sdist;   // h_dist
    __syncthreads();

    // e1 = relu(comb @ W1 + b1); thread tid owns output column tid
    float accv = b1[tid];
    for (int k = 0; k < COMB; ++k) accv += comb[k] * W1[(size_t)k * M_ + tid];
    e1[tid] = accv > 0.0f ? accv : 0.0f;
    __syncthreads();

    // preds = e1 @ W2 + b2
    for (int o = 0; o < 2; ++o) {
        red[tid] = e1[tid] * W2[tid * 2 + o];
        __syncthreads();
        for (int sft = 128; sft > 0; sft >>= 1) {
            if (tid < sft) red[tid] += red[tid + sft];
            __syncthreads();
        }
        if (tid == 0) out[b * 2 + o] = red[0] + b2[o];
        __syncthreads();
    }
}

// ---------------------------------------------------------------------------
extern "C" void kernel_launch(void* const* d_in, const int* in_sizes, int n_in,
                              void* d_out, int out_size, void* d_ws, size_t ws_size,
                              hipStream_t stream) {
    const int*   in1  = (const int*)d_in[0];
    const int*   in2  = (const int*)d_in[1];
    const int*   sl1  = (const int*)d_in[2];
    const int*   sl2  = (const int*)d_in[3];
    const float* emb  = (const float*)d_in[4];
    const float* kern = (const float*)d_in[5];
    const float* bias = (const float*)d_in[6];
    const float* W1   = (const float*)d_in[7];
    const float* b1   = (const float*)d_in[8];
    const float* W2   = (const float*)d_in[9];
    const float* b2   = (const float*)d_in[10];

    __bf16* frag = (__bf16*)d_ws;
    float*  hbuf = (float*)((char*)d_ws + (size_t)FRAG_TOTAL * 2);  // 2 x [B,H] f32

    hipLaunchKernelGGL(prep_wfrag, dim3((FRAG_TOTAL + 255) / 256), dim3(256), 0,
                       stream, kern, frag);
    hipLaunchKernelGGL(lstm_kernel, dim3(64), dim3(256), 0, stream,
                       in1, in2, sl1, sl2, emb, frag, bias, hbuf);
    hipLaunchKernelGGL(head_kernel, dim3(B_), dim3(256), 0, stream,
                       hbuf, sl1, sl2, W1, b1, W2, b2, (float*)d_out);
}
